// Network_68650757259527
// MI455X (gfx1250) — compile-verified, matrix operand order checked by reading, not run on hardware
//
#include <hip/hip_runtime.h>
#include <hip/hip_bf16.h>
#include <math.h>

#define NPTS 2048
#define BATCH 16
#define KNN 20
#define LRELU(v) ((v) > 0.f ? (v) : 0.2f * (v))

typedef __attribute__((ext_vector_type(16))) _Float16 v16h;
typedef __attribute__((ext_vector_type(8)))  _Float16 v8h;
typedef __attribute__((ext_vector_type(8)))  float    v8f;
typedef unsigned int u32x4 __attribute__((ext_vector_type(4)));
typedef int          i32x4 __attribute__((ext_vector_type(4)));
typedef int          i32x8 __attribute__((ext_vector_type(8)));

// Tensor Data Mover dispatch: therock header toolchain = 6-arg builtin,
// ROCm 7.2 clang-22 = 5-arg builtin, else fall back to async-to-LDS asm.
#if __has_include(<hip/amd_detail/amd_gfx1250_TDM.h>)
#define TDM_ARGS 6
#elif __has_builtin(__builtin_amdgcn_tensor_load_to_lds)
#define TDM_ARGS 5
#else
#define TDM_ARGS 0
#endif

__device__ __forceinline__ int lane_id() { return threadIdx.x & 31; }

// ---- CDNA5 async global->LDS (ASYNCcnt); generic addr low 32 bits == LDS offset
__device__ __forceinline__ void async_b128(void* lds, const void* g) {
  asm volatile("global_load_async_to_lds_b128 %0, %1, off"
               :: "v"((unsigned int)(uintptr_t)lds), "v"(g)
               : "memory");
}
__device__ __forceinline__ void async_wait0() {
  asm volatile("s_wait_asynccnt 0x0" ::: "memory");
}

__device__ __forceinline__ v8f wmma16(v16h a, v16h b, v8f c) {
  return __builtin_amdgcn_wmma_f32_16x16x32_f16(false, a, false, b, (short)0, c,
                                                false, false);
}

// 16x32 fragment from a row-major f16 matrix (pitch ld, LDS or global).
// Lane's 16 values are two contiguous 8-element chunks (ISA 7.12.2).
__device__ __forceinline__ v16h frag_rm(const _Float16* S, int ld, int row0, int k0) {
  const int l = lane_id(), half = l >> 4;
  const _Float16* p = S + (size_t)(row0 + (l & 15)) * ld + k0 + half * 8;
  v8h lo = *(const v8h*)p;
  v8h hi = *(const v8h*)(p + 16);
  v16h v;
#pragma unroll
  for (int j = 0; j < 8; ++j) { v[j] = lo[j]; v[j + 8] = hi[j]; }
  return v;
}

// store f32 C/D tile with per-row affine + leaky-relu as f16 into a
// [col][feat] buffer (pitch ld): one contiguous 16B store per lane.
__device__ __forceinline__ void store_d_lrelu(_Float16* S, int ld, int row0, int col0,
                                              v8f acc, const float* sc, const float* bs) {
  const int l = lane_id(), half = l >> 4, n = col0 + (l & 15);
  const int rbase = row0 + 8 * half;
  v8h o;
#pragma unroll
  for (int i = 0; i < 8; ++i) {
    int r = rbase + i;
    float y = acc[i] * sc[r] + bs[r];
    o[i] = (_Float16)LRELU(y);
  }
  *(v8h*)(S + (size_t)n * ld + rbase) = o;
}

// ------------------------- fp32 -> zero-padded f16 weight conversion (once)
__global__ void __launch_bounds__(256) cvt_pad_kernel(
    const float* __restrict__ src, _Float16* __restrict__ dst,
    int srcLd, int colOff, int R, int C, int Cp, int total) {
  int t = blockIdx.x * 256 + threadIdx.x;
  if (t >= total) return;
  int r = t / Cp, c = t % Cp;
  float v = 0.f;
  if (r < R && c < C) v = src[(size_t)r * srcLd + colOff + c];
  dst[t] = (_Float16)v;
}

// ---------------------------------------------------------------- knn (top-20)
template <int C>
__global__ void __launch_bounds__(256) knn_kernel(const float* __restrict__ x,
                                                  int* __restrict__ idx) {
  const int b = blockIdx.y;
  const int n = blockIdx.x * 256 + threadIdx.x;
  float q[C];
#pragma unroll
  for (int c = 0; c < C; ++c) q[c] = x[(b * C + c) * NPTS + n];
  float qq = 0.f;
#pragma unroll
  for (int c = 0; c < C; ++c) qq += q[c] * q[c];

  float bd[KNN]; int bi[KNN];
#pragma unroll
  for (int k = 0; k < KNN; ++k) { bd[k] = -1e30f; bi[k] = 0; }

  const int TILE = 128;
  __shared__ __align__(16) float tile[C][TILE];
  __shared__ float txx[TILE];

  for (int t0 = 0; t0 < NPTS; t0 += TILE) {
    __syncthreads();
    // async DMA candidate tile straight into LDS (bypasses VGPRs)
    for (int t = threadIdx.x; t < C * (TILE / 4); t += 256) {
      int c = t / (TILE / 4), jj = t % (TILE / 4);
      async_b128(&tile[c][jj * 4], x + (b * C + c) * NPTS + t0 + jj * 4);
    }
    if (t0 + TILE < NPTS)  // hint next tile toward the caches
      __builtin_prefetch(x + (b * C) * NPTS + t0 + TILE + threadIdx.x, 0, 0);
    async_wait0();
    __syncthreads();
    if (threadIdx.x < TILE) {
      float s = 0.f;
#pragma unroll
      for (int c = 0; c < C; ++c) { float v = tile[c][threadIdx.x]; s += v * v; }
      txx[threadIdx.x] = s;
    }
    __syncthreads();
    for (int j = 0; j < TILE; ++j) {
      float dot = 0.f;
#pragma unroll
      for (int c = 0; c < C; ++c) dot += q[c] * tile[c][j];
      float d = 2.f * dot - qq - txx[j];
      if (d > bd[KNN - 1]) {
        int pos = KNN - 1;
        while (pos > 0 && bd[pos - 1] < d) {
          bd[pos] = bd[pos - 1]; bi[pos] = bi[pos - 1]; --pos;
        }
        bd[pos] = d; bi[pos] = t0 + j;
      }
    }
  }
  for (int k = 0; k < KNN; ++k) idx[(b * NPTS + n) * KNN + k] = bi[k];
}

// ------------------------------------------- fused EdgeConv (gather+GEMMs+kmax)
template <int CIN, int TWO>
__global__ void __launch_bounds__(256) edgeconv_kernel(
    const float* __restrict__ x, const int* __restrict__ idx,
    const _Float16* __restrict__ w1h, const float* __restrict__ s1g, const float* __restrict__ b1g,
    const _Float16* __restrict__ w2h, const float* __restrict__ s2g, const float* __restrict__ b2g,
    float* __restrict__ xout, _Float16* __restrict__ xc, int rowOff) {
  constexpr int F  = 2 * CIN;
  constexpr int FP = (F + 31) & ~31;
  constexpr int PTS = 8;
  constexpr int COLS = PTS * KNN;   // 160 = 10 col tiles
  constexpr int NCT = COLS / 16;
  const int b = blockIdx.y;
  const int nbase = blockIdx.x * PTS;
  const int tid = threadIdx.x;
  const int wave = tid >> 5;

  __shared__ __align__(16) _Float16 sFeat[COLS][FP];
  __shared__ __align__(16) _Float16 sMid[COLS][64];
  __shared__ __align__(16) _Float16 sOut[COLS][64];
  __shared__ __align__(16) _Float16 sW1[64][FP];
  __shared__ __align__(16) _Float16 sW2[64][64];
  __shared__ float sS1[64], sB1[64], sS2[64], sB2[64];
  __shared__ int sNbr[COLS];

  // async-DMA weights into LDS + affine params + neighbor indices
  for (int t = tid; t < 64 * FP / 8; t += 256)
    async_b128(&((uint4*)&sW1[0][0])[t], &((const uint4*)w1h)[t]);
  if (TWO)
    for (int t = tid; t < 64 * 64 / 8; t += 256)
      async_b128(&((uint4*)&sW2[0][0])[t], &((const uint4*)w2h)[t]);
  if (tid < 64) {
    sS1[tid] = s1g[tid]; sB1[tid] = b1g[tid];
    if (TWO) { sS2[tid] = s2g[tid]; sB2[tid] = b2g[tid]; }
  }
  if (tid < COLS) {
    int n = nbase + tid / KNN;
    sNbr[tid] = idx[(b * NPTS + n) * KNN + (tid % KNN)];
  }
  async_wait0();
  __syncthreads();

  // gather edge features: [nbr - ctr ; ctr], stored [col][feat]
  for (int t = tid; t < COLS * FP; t += 256) {
    int c = t / FP, f = t % FP;
    int n = nbase + c / KNN;
    float v = 0.f;
    if (f < CIN) {
      int j = sNbr[c];
      v = x[(b * CIN + f) * NPTS + j] - x[(b * CIN + f) * NPTS + n];
    } else if (f < F) {
      v = x[(b * CIN + (f - CIN)) * NPTS + n];
    }
    sFeat[c][f] = (_Float16)v;
  }
  __syncthreads();

  // layer 1: [64 x F] * [F x 160]
  for (int job = wave; job < 4 * NCT; job += 8) {
    int rt = job / NCT, ct = job % NCT;
    v8f acc = {};
#pragma unroll
    for (int k0 = 0; k0 < FP; k0 += 32) {
      v16h a  = frag_rm(&sW1[0][0], FP, rt * 16, k0);
      v16h bb = frag_rm(&sFeat[0][0], FP, ct * 16, k0);
      acc = wmma16(a, bb, acc);
    }
    store_d_lrelu(&sMid[0][0], 64, rt * 16, ct * 16, acc, sS1, sB1);
  }
  __syncthreads();

  if (TWO) {
    for (int job = wave; job < 4 * NCT; job += 8) {
      int rt = job / NCT, ct = job % NCT;
      v8f acc = {};
#pragma unroll
      for (int k0 = 0; k0 < 64; k0 += 32) {
        v16h a  = frag_rm(&sW2[0][0], 64, rt * 16, k0);
        v16h bb = frag_rm(&sMid[0][0], 64, ct * 16, k0);
        acc = wmma16(a, bb, acc);
      }
      store_d_lrelu(&sOut[0][0], 64, rt * 16, ct * 16, acc, sS2, sB2);
    }
    __syncthreads();
  }

  // max over the 20 neighbor columns of each point
  const _Float16* src = TWO ? &sOut[0][0] : &sMid[0][0];
  for (int t = tid; t < 64 * PTS; t += 256) {
    int r = t & 63, p = t >> 6;
    float m = -1e30f;
#pragma unroll
    for (int k = 0; k < KNN; ++k)
      m = fmaxf(m, (float)src[(p * KNN + k) * 64 + r]);
    int n = nbase + p;
    if (xout) xout[(b * 64 + r) * NPTS + n] = m;
    xc[((size_t)(b * NPTS + n)) * 192 + rowOff + r] = (_Float16)m;
  }
}

// --------------------------- conv6 (1024x192) + global max over N, fused
__global__ void __launch_bounds__(256) conv6max_kernel(
    const _Float16* __restrict__ xc, const _Float16* __restrict__ w6h,
    const float* __restrict__ s6, const float* __restrict__ b6,
    float* __restrict__ g) {
  const int b = blockIdx.y;
  const int row0 = blockIdx.x * 64;
  const int tid = threadIdx.x;
  const int wave = tid >> 5;
  const int rs = wave & 3;
  const int cpart = wave >> 2;
  const int l = tid & 31, half = l >> 4;

  __shared__ float red[2][64];
  __shared__ float sS[64], sB[64];
  if (tid < 64) { sS[tid] = s6[row0 + tid]; sB[tid] = b6[row0 + tid]; }
  __syncthreads();

  const _Float16* xcb = xc + (size_t)b * NPTS * 192;

  v16h afrag[6];
#pragma unroll
  for (int kk = 0; kk < 6; ++kk)
    afrag[kk] = frag_rm(w6h, 192, row0 + rs * 16, kk * 32);

  v8f rmax;
#pragma unroll
  for (int i = 0; i < 8; ++i) rmax[i] = -1e30f;

  for (int ct = cpart; ct < NPTS / 16; ct += 2) {
    v8f acc = {};
#pragma unroll
    for (int kk = 0; kk < 6; ++kk) {
      v16h bb = frag_rm(xcb, 192, ct * 16, kk * 32);
      acc = wmma16(afrag[kk], bb, acc);
    }
#pragma unroll
    for (int i = 0; i < 8; ++i) {
      int r = rs * 16 + i + 8 * half;
      float y = acc[i] * sS[r] + sB[r];
      rmax[i] = fmaxf(rmax[i], LRELU(y));
    }
  }
#pragma unroll
  for (int m = 1; m <= 8; m <<= 1)
#pragma unroll
    for (int i = 0; i < 8; ++i)
      rmax[i] = fmaxf(rmax[i], __shfl_xor(rmax[i], m, 32));

  if ((l & 15) == 0) {
#pragma unroll
    for (int i = 0; i < 8; ++i) red[cpart][rs * 16 + i + 8 * half] = rmax[i];
  }
  __syncthreads();
  if (tid < 64)
    g[b * 1024 + row0 + tid] = fmaxf(red[0][tid], red[1][tid]);
}

// ------------- per-batch constant part of MLP L1: wf1[:, :1088] * [g ; cv]
__global__ void __launch_bounds__(256) const1_kernel(
    const float* __restrict__ cls, const float* __restrict__ wc,
    const float* __restrict__ scg, const float* __restrict__ bcg,
    const float* __restrict__ wf1, const float* __restrict__ g,
    float* __restrict__ c1) {
  const int b = blockIdx.x;
  __shared__ float cv[64];
  if (threadIdx.x < 64) {
    float a = 0.f;
    for (int c = 0; c < 16; ++c) a += wc[threadIdx.x * 16 + c] * cls[b * 16 + c];
    a = a * scg[threadIdx.x] + bcg[threadIdx.x];
    cv[threadIdx.x] = LRELU(a);
  }
  __syncthreads();
  const int o = threadIdx.x;
  const float* wr = wf1 + (size_t)o * 1280;
  float acc = 0.f;
  for (int i = 0; i < 1024; ++i) acc += wr[i] * g[b * 1024 + i];
  for (int j = 0; j < 64; ++j)  acc += wr[1024 + j] * cv[j];
  c1[b * 256 + o] = acc;
}

// ------------- fused 4-layer point MLP + log_softmax, 64 columns / block
__global__ void __launch_bounds__(256) mlp_kernel(
    const _Float16* __restrict__ xc, const float* __restrict__ c1,
    const _Float16* __restrict__ wf1h, const float* __restrict__ sf1, const float* __restrict__ bf1,
    const _Float16* __restrict__ wf2h, const float* __restrict__ sf2, const float* __restrict__ bf2,
    const _Float16* __restrict__ wf3h, const float* __restrict__ sf3, const float* __restrict__ bf3,
    const _Float16* __restrict__ wf4h, float* __restrict__ out) {
  const int b = blockIdx.y;
  const int nb = blockIdx.x * 64;
  const int tid = threadIdx.x;
  const int wave = tid >> 5;
  const int l = tid & 31, half = l >> 4;

  __shared__ __align__(16) _Float16 sIn[64][192];
  __shared__ __align__(16) _Float16 sA[64][256];
  __shared__ __align__(16) _Float16 sB[64][256];
  __shared__ __align__(16) _Float16 sC[64][128];
  __shared__ float sLog[64][64];        // [col][row]
  __shared__ float sC1[256];
  __shared__ float aS1[256], aB1[256], aS2[256], aB2[256], aS3[128], aB3[128];

  const _Float16* gsrc = xc + (size_t)(b * NPTS + nb) * 192;
#if TDM_ARGS > 0
  // Tensor Data Mover: one descriptor moves the whole 64x192 f16 tile.
  if (wave == 0) {
    unsigned int lds = (unsigned int)(uintptr_t)&sIn[0][0];
    unsigned long long ga = (unsigned long long)(uintptr_t)gsrc;
    u32x4 g0 = { 1u,                                   // count=1 valid descriptor
                 lds,                                   // lds_addr
                 (unsigned int)ga,                      // global_addr[31:0]
                 (unsigned int)((ga >> 32) & 0x01FFFFFFu) | (2u << 30) }; // [56:32]|type=2
    i32x8 g1 = { (int)0x00010000,        // data_size=2B, no multicast/pad/iterate
                 (int)(192u << 16),      // atomic_barrier=0 | tensor_dim0=192 (lo)
                 (int)(64u << 16),       // tensor_dim0 hi=0 | tensor_dim1=64 (lo)
                 (int)(192u << 16),      // tensor_dim1 hi=0 | tile_dim0=192
                 (int)64,                // tile_dim1=64, tile_dim2=0
                 (int)192,               // tensor_dim0_stride=192 (lo)
                 0, 0 };                 // stride hi, tensor_dim1_stride=0
    i32x4 z4 = {0, 0, 0, 0};
#if TDM_ARGS == 6
    i32x8 z8 = {0, 0, 0, 0, 0, 0, 0, 0};
    __builtin_amdgcn_tensor_load_to_lds(g0, g1, z4, z4, z8, 0);
#else
    __builtin_amdgcn_tensor_load_to_lds(g0, g1, z4, z4, 0);
#endif
    __builtin_amdgcn_s_wait_tensorcnt(0);
  }
#else
  for (int t = tid; t < 64 * 192 / 8; t += 256)
    async_b128(&((uint4*)&sIn[0][0])[t], &((const uint4*)gsrc)[t]);
  async_wait0();
#endif
  sC1[tid] = c1[b * 256 + tid];
  aS1[tid] = sf1[tid]; aB1[tid] = bf1[tid];
  aS2[tid] = sf2[tid]; aB2[tid] = bf2[tid];
  if (tid < 128) { aS3[tid] = sf3[tid]; aB3[tid] = bf3[tid]; }
  __syncthreads();

  // L1: 256 x 192 variable part + per-batch constant, affine, lrelu
  for (int job = wave; job < 64; job += 8) {
    int rt = job >> 2, ct = job & 3;
    v8f acc = {};
#pragma unroll
    for (int k0 = 0; k0 < 192; k0 += 32) {
      v16h a  = frag_rm(wf1h, 192, rt * 16, k0);
      v16h bb = frag_rm(&sIn[0][0], 192, ct * 16, k0);
      acc = wmma16(a, bb, acc);
    }
    const int n = ct * 16 + (l & 15), rbase = rt * 16 + 8 * half;
    v8h o;
#pragma unroll
    for (int i = 0; i < 8; ++i) {
      int r = rbase + i;
      float y = (acc[i] + sC1[r]) * aS1[r] + aB1[r];
      o[i] = (_Float16)LRELU(y);
    }
    *(v8h*)(&sA[n][rbase]) = o;
  }
  __syncthreads();
  // L2: 256x256
  for (int job = wave; job < 64; job += 8) {
    int rt = job >> 2, ct = job & 3;
    v8f acc = {};
#pragma unroll
    for (int k0 = 0; k0 < 256; k0 += 32) {
      v16h a  = frag_rm(wf2h, 256, rt * 16, k0);
      v16h bb = frag_rm(&sA[0][0], 256, ct * 16, k0);
      acc = wmma16(a, bb, acc);
    }
    store_d_lrelu(&sB[0][0], 256, rt * 16, ct * 16, acc, aS2, aB2);
  }
  __syncthreads();
  // L3: 128x256
  for (int job = wave; job < 32; job += 8) {
    int rt = job >> 2, ct = job & 3;
    v8f acc = {};
#pragma unroll
    for (int k0 = 0; k0 < 256; k0 += 32) {
      v16h a  = frag_rm(wf3h, 256, rt * 16, k0);
      v16h bb = frag_rm(&sB[0][0], 256, ct * 16, k0);
      acc = wmma16(a, bb, acc);
    }
    store_d_lrelu(&sC[0][0], 128, rt * 16, ct * 16, acc, aS3, aB3);
  }
  __syncthreads();
  // L4: 50x128 (rows zero-padded to 64 in wf4h), raw logits
  for (int job = wave; job < 16; job += 8) {
    int rt = job >> 2, ct = job & 3;
    v8f acc = {};
#pragma unroll
    for (int k0 = 0; k0 < 128; k0 += 32) {
      v16h a  = frag_rm(wf4h, 128, rt * 16, k0);
      v16h bb = frag_rm(&sC[0][0], 128, ct * 16, k0);
      acc = wmma16(a, bb, acc);
    }
    const int n = ct * 16 + (l & 15), rbase = rt * 16 + 8 * half;
#pragma unroll
    for (int i = 0; i < 8; ++i) sLog[n][rbase + i] = acc[i];
  }
  __syncthreads();
  if (tid < 64) {
    const int c = tid;
    float m = -1e30f;
    for (int r = 0; r < 50; ++r) m = fmaxf(m, sLog[c][r]);
    float s = 0.f;
    for (int r = 0; r < 50; ++r) s += __expf(sLog[c][r] - m);
    float lse = m + __logf(s);
    for (int r = 0; r < 50; ++r)
      out[((size_t)b * NPTS + nb + c) * 50 + r] = sLog[c][r] - lse;
  }
}

// ---------------------------------------------------------------- launcher
extern "C" void kernel_launch(void* const* d_in, const int* in_sizes, int n_in,
                              void* d_out, int out_size, void* d_ws, size_t ws_size,
                              hipStream_t stream) {
  (void)in_sizes; (void)n_in; (void)out_size; (void)ws_size;
  const float* points = (const float*)d_in[0];
  const float* clsl   = (const float*)d_in[1];
  const float* w1 = (const float*)d_in[2];  const float* s1 = (const float*)d_in[3];  const float* b1 = (const float*)d_in[4];
  const float* w2 = (const float*)d_in[5];  const float* s2 = (const float*)d_in[6];  const float* b2 = (const float*)d_in[7];
  const float* w3 = (const float*)d_in[8];  const float* s3 = (const float*)d_in[9];  const float* b3 = (const float*)d_in[10];
  const float* w4 = (const float*)d_in[11]; const float* s4 = (const float*)d_in[12]; const float* b4 = (const float*)d_in[13];
  const float* w5 = (const float*)d_in[14]; const float* s5 = (const float*)d_in[15]; const float* b5 = (const float*)d_in[16];
  const float* w6 = (const float*)d_in[17]; const float* s6 = (const float*)d_in[18]; const float* b6 = (const float*)d_in[19];
  const float* wc = (const float*)d_in[20]; const float* sc = (const float*)d_in[21]; const float* bc = (const float*)d_in[22];
  const float* wf1 = (const float*)d_in[23]; const float* sf1 = (const float*)d_in[24]; const float* bf1 = (const float*)d_in[25];
  const float* wf2 = (const float*)d_in[26]; const float* sf2 = (const float*)d_in[27]; const float* bf2 = (const float*)d_in[28];
  const float* wf3 = (const float*)d_in[29]; const float* sf3 = (const float*)d_in[30]; const float* bf3 = (const float*)d_in[31];
  const float* wf4 = (const float*)d_in[32];

  char* p = (char*)d_ws;
  auto carve = [&](size_t bytes) -> void* {
    void* r = p;
    p += (bytes + 255) & ~(size_t)255;
    return r;
  };
  int*      idx  = (int*)     carve((size_t)BATCH * NPTS * KNN * sizeof(int));
  float*    x1   = (float*)   carve((size_t)BATCH * 64 * NPTS * sizeof(float));
  float*    x2   = (float*)   carve((size_t)BATCH * 64 * NPTS * sizeof(float));
  _Float16* xc   = (_Float16*)carve((size_t)BATCH * NPTS * 192 * sizeof(_Float16)); // [B][N][192]
  float*    g    = (float*)   carve((size_t)BATCH * 1024 * sizeof(float));
  float*    c1   = (float*)   carve((size_t)BATCH * 256 * sizeof(float));
  _Float16* w1h  = (_Float16*)carve(64 * 32 * 2);
  _Float16* w2h  = (_Float16*)carve(64 * 64 * 2);
  _Float16* w3h  = (_Float16*)carve(64 * 128 * 2);
  _Float16* w4h  = (_Float16*)carve(64 * 64 * 2);
  _Float16* w5h  = (_Float16*)carve(64 * 128 * 2);
  _Float16* w6h  = (_Float16*)carve(1024 * 192 * 2);
  _Float16* wf1h = (_Float16*)carve(256 * 192 * 2);
  _Float16* wf2h = (_Float16*)carve(256 * 256 * 2);
  _Float16* wf3h = (_Float16*)carve(128 * 256 * 2);
  _Float16* wf4h = (_Float16*)carve(64 * 128 * 2);

  dim3 blk(256);
  // conversions (R rows kept, rest zero; Cp = padded col count = dst pitch)
  cvt_pad_kernel<<<dim3((64 * 32 + 255) / 256), blk, 0, stream>>>(w1, w1h, 6, 0, 64, 6, 32, 64 * 32);
  cvt_pad_kernel<<<dim3((64 * 64 + 255) / 256), blk, 0, stream>>>(w2, w2h, 64, 0, 64, 64, 64, 64 * 64);
  cvt_pad_kernel<<<dim3((64 * 128 + 255) / 256), blk, 0, stream>>>(w3, w3h, 128, 0, 64, 128, 128, 64 * 128);
  cvt_pad_kernel<<<dim3((64 * 64 + 255) / 256), blk, 0, stream>>>(w4, w4h, 64, 0, 64, 64, 64, 64 * 64);
  cvt_pad_kernel<<<dim3((64 * 128 + 255) / 256), blk, 0, stream>>>(w5, w5h, 128, 0, 64, 128, 128, 64 * 128);
  cvt_pad_kernel<<<dim3((1024 * 192 + 255) / 256), blk, 0, stream>>>(w6, w6h, 192, 0, 1024, 192, 192, 1024 * 192);
  cvt_pad_kernel<<<dim3((256 * 192 + 255) / 256), blk, 0, stream>>>(wf1, wf1h, 1280, 1088, 256, 192, 192, 256 * 192);
  cvt_pad_kernel<<<dim3((256 * 256 + 255) / 256), blk, 0, stream>>>(wf2, wf2h, 256, 0, 256, 256, 256, 256 * 256);
  cvt_pad_kernel<<<dim3((128 * 256 + 255) / 256), blk, 0, stream>>>(wf3, wf3h, 256, 0, 128, 256, 256, 128 * 256);
  cvt_pad_kernel<<<dim3((64 * 128 + 255) / 256), blk, 0, stream>>>(wf4, wf4h, 128, 0, 50, 128, 128, 64 * 128);

  // block 1
  knn_kernel<3><<<dim3(NPTS / 256, BATCH), blk, 0, stream>>>(points, idx);
  edgeconv_kernel<3, 1><<<dim3(NPTS / 8, BATCH), blk, 0, stream>>>(
      points, idx, w1h, s1, b1, w2h, s2, b2, x1, xc, 0);
  // block 2
  knn_kernel<64><<<dim3(NPTS / 256, BATCH), blk, 0, stream>>>(x1, idx);
  edgeconv_kernel<64, 1><<<dim3(NPTS / 8, BATCH), blk, 0, stream>>>(
      x1, idx, w3h, s3, b3, w4h, s4, b4, x2, xc, 64);
  // block 3 (single conv)
  knn_kernel<64><<<dim3(NPTS / 256, BATCH), blk, 0, stream>>>(x2, idx);
  edgeconv_kernel<64, 0><<<dim3(NPTS / 8, BATCH), blk, 0, stream>>>(
      x2, idx, w5h, s5, b5, nullptr, nullptr, nullptr, nullptr, xc, 128);
  // global feature + per-batch constant MLP part
  conv6max_kernel<<<dim3(1024 / 64, BATCH), blk, 0, stream>>>(xc, w6h, s6, b6, g);
  const1_kernel<<<dim3(BATCH), blk, 0, stream>>>(clsl, wc, sc, bc, wf1, g, c1);
  // fused point MLP + log_softmax
  mlp_kernel<<<dim3(NPTS / 64, BATCH), blk, 0, stream>>>(
      xc, c1, wf1h, sf1, bf1, wf2h, sf2, bf2, wf3h, sf3, bf3, wf4h, (float*)d_out);
}
